// pointConv_88175678587631
// MI455X (gfx1250) — matrix operand order repaired and checked by reading, not compiled, not run on hardware
//
#include <hip/hip_runtime.h>

// ---------------------------------------------------------------------------
// pointConv on gfx1250 via one-hot WMMA + Tensor Data Mover staging:
//   out[i, cell, c] = mean_{j in cell(i)} attr[j, c]
//   per center i: D(16x64) = OneHot_i(16x8192) * Attr(8192x64); counts via an
//   extra WMMA against an all-ones B tile. f16 inputs, f32 accumulate.
//   B-panel chunks (4KB) are DMA'd into double-buffered LDS by the TDM and
//   shared by all 8 waves of the block (8x VMEM traffic reduction).
// ---------------------------------------------------------------------------

typedef _Float16     v16h __attribute__((ext_vector_type(16)));
typedef float        v8f  __attribute__((ext_vector_type(8)));
typedef unsigned int v4u  __attribute__((ext_vector_type(4)));
typedef int          v8i  __attribute__((ext_vector_type(8)));
typedef int          v4i  __attribute__((ext_vector_type(4)));

#define N_POINTS 8192
#define CHANNELS 64
#define NUM_CELLS 16
#define CHUNK 32
#define NCHUNK (N_POINTS / CHUNK)   // 256
#define CHUNK_BYTES 4096            // 4 ct * 32 lanes * 32 B

#if defined(__has_builtin)
#  if __has_builtin(__builtin_amdgcn_tensor_load_to_lds)
#    define USE_TDM 1
#  endif
#endif
#ifndef USE_TDM
#  define USE_TDM 0
#endif

#if __has_include(<hip/amd_detail/amd_gfx1250_TDM.h>)
#  define TDM_6ARG 1
#else
#  define TDM_6ARG 0
#endif

union B32x8 {
    unsigned int u[8];
    v4u          q[2];
    v16h         h;
};

// --- prep: f32 attributes -> f16 "B-panel" in WMMA B-operand layout ---------
// panel dword t = ((jc*4 + ct)*32 + L)*8 + v holds packed f16 pair
//   lo = attr[jb][c], hi = attr[jb+1][c]
//   with c = ct*16 + (L&15), jb = jc*32 + (L>>4)*16 + 2v
// B in this code's contiguous k order (not AMD's table order), as A: VGPR v, lanes 0-15 hold
// K = 2v,2v+1 ; lanes 16-31 hold K = 16+2v,17+2v ; N = lane&15.
__global__ void prep_panel_kernel(const float* __restrict__ attr,
                                  _Float16* __restrict__ panel) {
    int t    = blockIdx.x * blockDim.x + threadIdx.x;  // exact grid, no bounds
    int v    = t & 7;
    int L    = (t >> 3) & 31;
    int ctjc = t >> 8;
    int ct   = ctjc & 3;
    int jc   = ctjc >> 2;
    int c    = ct * 16 + (L & 15);
    int jb   = jc * CHUNK + ((L >> 4) * 16) + 2 * v;
    panel[2 * t + 0] = (_Float16)attr[jb * CHANNELS + c];
    panel[2 * t + 1] = (_Float16)attr[(jb + 1) * CHANNELS + c];
}

#if USE_TDM
// Issue a TDM DMA of one 4KB panel chunk into LDS (wave-uniform values).
// D# built per CDNA5 ISA 8.3-8.6: 1-row tensor of 1024 dwords.
__device__ __forceinline__ void tdm_load_chunk(const _Float16* panel, int jc,
                                               unsigned ldsOff) {
    unsigned long long ga =
        (unsigned long long)(uintptr_t)panel + (unsigned long long)jc * CHUNK_BYTES;
    v4u g0;
    g0[0] = 1u;                                        // count=1 (user mode)
    g0[1] = ldsOff;                                    // lds_addr [63:32]
    g0[2] = (unsigned)(ga & 0xffffffffu);              // global_addr lo
    g0[3] = (unsigned)((ga >> 32) & 0x01ffffffu)       // global_addr hi (57b)
            | 0x80000000u;                             // type=2 ("image")
    v8i g1;
    g1[0] = 0x00020000;  // wg_mask=0, data_size=2 (4B), no flags
    g1[1] = 0x04000000;  // tensor_dim0 = 1024 (bits 63:48)
    g1[2] = 0x00010000;  // tensor_dim0 hi=0, tensor_dim1 = 1
    g1[3] = 0x04000000;  // tile_dim0 = 1024 (bits 127:112)
    g1[4] = 1;           // tile_dim1 = 1, tile_dim2 = 0
    g1[5] = 1024;        // tensor_dim0_stride lo
    g1[6] = 0;           // stride hi, tensor_dim1_stride lo
    g1[7] = 0;
    v4i gz = {0, 0, 0, 0};
#if TDM_6ARG
    v8i gz8 = {0, 0, 0, 0, 0, 0, 0, 0};
    __builtin_amdgcn_tensor_load_to_lds(g0, g1, gz, gz, gz8, 0);
#else
    __builtin_amdgcn_tensor_load_to_lds(g0, g1, gz, gz, 0);
#endif
}

__device__ __forceinline__ void wait_tensorcnt0() {
#if __has_builtin(__builtin_amdgcn_s_wait_tensorcnt)
    __builtin_amdgcn_s_wait_tensorcnt(0);
#else
    asm volatile("s_wait_tensorcnt 0x0" ::: "memory");
#endif
}
#endif  // USE_TDM

// --- main: one wave per center point, 8 waves/block share LDS panels --------
__global__ void __launch_bounds__(256)
pointconv_wmma_kernel(const float* __restrict__ pc,
                      const _Float16* __restrict__ panel,
                      float* __restrict__ out) {
    __shared__ __align__(16) unsigned char ldsBuf[2][CHUNK_BYTES];

    const int t    = threadIdx.x;
    const int lane = t & 31;
    const int wave = t >> 5;
    const int i    = blockIdx.x * 8 + wave;

    const float xi = pc[3 * i + 0];
    const float yi = pc[3 * i + 1];
    const float zi = pc[3 * i + 2];

    v8f acc0 = {}, acc1 = {}, acc2 = {}, acc3 = {};
    v8f cntAcc = {};

    B32x8 bOnes;
#pragma unroll
    for (int v = 0; v < 8; ++v) bOnes.u[v] = 0x3C003C00u;  // 1.0h pairs

    const bool hiHalf = (lane & 16) != 0;
    const int  m      = lane & 15;  // this lane's cell row in the A operand

#if !USE_TDM
    const v4u* __restrict__ panel4 = (const v4u*)panel;
#endif
    const unsigned ldsBase = (unsigned)(uintptr_t)&ldsBuf[0][0];
    (void)ldsBase;

    // --- prologue: stage chunk 0 into buffer 0 ------------------------------
#if USE_TDM
    if (wave == 0) tdm_load_chunk(panel, 0, ldsBase);
#else
    {
        v4u tmp = ((const v4u*)panel)[t];
        *(v4u*)&ldsBuf[0][t * 16] = tmp;
    }
#endif

    for (int jc = 0; jc < NCHUNK; ++jc) {
#if USE_TDM
        if (wave == 0) wait_tensorcnt0();     // buf[jc&1] DMA complete
#endif
        __syncthreads();                      // publish buffer to all waves

        // kick off next chunk's DMA into the other buffer
        if (jc + 1 < NCHUNK) {
#if USE_TDM
            if (wave == 0)
                tdm_load_chunk(panel, jc + 1,
                               ldsBase + (unsigned)(((jc + 1) & 1) * CHUNK_BYTES));
#else
            v4u tmp = ((const v4u*)panel)[(size_t)(jc + 1) * 256 + t];
            *(v4u*)&ldsBuf[(jc + 1) & 1][t * 16] = tmp;
#endif
            __builtin_prefetch(&pc[3 * ((jc + 1) * CHUNK + lane)], 0, 1);
        }

        // --- lane computes the cell of neighbor j = jc*32 + lane -----------
        int   j  = jc * CHUNK + lane;
        float dx = pc[3 * j + 0] - xi;
        float dy = pc[3 * j + 1] - yi;
        float dz = pc[3 * j + 2] - zi;
        float d2 = dx * dx + dy * dy + dz * dz;
        int cell = (dx >= 0.f ? 4 : 0) | (dy >= 0.f ? 2 : 0) | (dz >= 0.f ? 1 : 0);
        if (d2 >= 0.0025f) cell |= 8;          // shell 1 (dist >= 0.05)
        cell = (d2 <= 0.01f) ? cell : 255;     // outside radius -> no cell

        // --- build one-hot A tile (16 cells x 32 neighbors, f16) -----------
        // A layout: lane m(0-15): VGPR v -> K=2v,2v+1
        //           lane 16+m   : same K indices + 16
        B32x8 aT;
#pragma unroll
        for (int v = 0; v < 8; ++v) {
            const int kL  = 2 * v;
            int lo0 = __builtin_amdgcn_readlane(cell, kL);
            int lo1 = __builtin_amdgcn_readlane(cell, kL + 1);
            int hi0 = __builtin_amdgcn_readlane(cell, kL + 16);
            int hi1 = __builtin_amdgcn_readlane(cell, kL + 17);
            int c0  = hiHalf ? hi0 : lo0;
            int c1  = hiHalf ? hi1 : lo1;
            aT.u[v] = (c0 == m ? 0x00003C00u : 0u) | (c1 == m ? 0x3C000000u : 0u);
        }

        // counts: OneHot x Ones -> per-cell neighbor count (replicated cols)
        cntAcc = __builtin_amdgcn_wmma_f32_16x16x32_f16(
            false, aT.h, false, bOnes.h, (short)0, cntAcc, false, false);

        // --- 4 channel tiles from LDS: sums += OneHot x Attr ---------------
        const unsigned char* cbuf = &ldsBuf[jc & 1][0];
        B32x8 b0, b1, b2, b3;
        {
            const v4u* lp0 = (const v4u*)(cbuf + (0 * 32 + lane) * 32);
            const v4u* lp1 = (const v4u*)(cbuf + (1 * 32 + lane) * 32);
            const v4u* lp2 = (const v4u*)(cbuf + (2 * 32 + lane) * 32);
            const v4u* lp3 = (const v4u*)(cbuf + (3 * 32 + lane) * 32);
            b0.q[0] = lp0[0]; b0.q[1] = lp0[1];
            b1.q[0] = lp1[0]; b1.q[1] = lp1[1];
            b2.q[0] = lp2[0]; b2.q[1] = lp2[1];
            b3.q[0] = lp3[0]; b3.q[1] = lp3[1];
        }
        acc0 = __builtin_amdgcn_wmma_f32_16x16x32_f16(
            false, aT.h, false, b0.h, (short)0, acc0, false, false);
        acc1 = __builtin_amdgcn_wmma_f32_16x16x32_f16(
            false, aT.h, false, b1.h, (short)0, acc1, false, false);
        acc2 = __builtin_amdgcn_wmma_f32_16x16x32_f16(
            false, aT.h, false, b2.h, (short)0, acc2, false, false);
        acc3 = __builtin_amdgcn_wmma_f32_16x16x32_f16(
            false, aT.h, false, b3.h, (short)0, acc3, false, false);
    }

    // --- divide by counts and store -----------------------------------------
    // D layout: VGPR r, lanes 0-15 -> row M=r ; lanes 16-31 -> row M=8+r ;
    // column N = lane&15. Count tile has counts replicated over columns.
    float* outp = out + (size_t)i * (NUM_CELLS * CHANNELS);
#pragma unroll
    for (int r = 0; r < 8; ++r) {
        int   mrow = r + (hiHalf ? 8 : 0);
        float inv  = 1.0f / fmaxf(cntAcc[r], 1.0f);
        outp[mrow * CHANNELS + 0 * 16 + m] = acc0[r] * inv;
        outp[mrow * CHANNELS + 1 * 16 + m] = acc1[r] * inv;
        outp[mrow * CHANNELS + 2 * 16 + m] = acc2[r] * inv;
        outp[mrow * CHANNELS + 3 * 16 + m] = acc3[r] * inv;
    }
}

extern "C" void kernel_launch(void* const* d_in, const int* in_sizes, int n_in,
                              void* d_out, int out_size, void* d_ws, size_t ws_size,
                              hipStream_t stream) {
    const float* attr  = (const float*)d_in[0];  // [8192, 64] f32
    const float* pc    = (const float*)d_in[1];  // [8192, 3]  f32
    float*       out   = (float*)d_out;          // [8192, 16, 64] f32
    _Float16*    panel = (_Float16*)d_ws;        // 1 MB f16 B-panel

    // 262144 panel dwords -> 1024 blocks x 256 threads (exact)
    prep_panel_kernel<<<1024, 256, 0, stream>>>(attr, panel);
    // one wave per point: 8192 waves = 1024 blocks x 8 waves
    pointconv_wmma_kernel<<<1024, 256, 0, stream>>>(pc, panel, out);
}
